// MergeModuleTemporal_13812614824518
// MI455X (gfx1250) — matrix operand order repaired and checked by reading, not compiled, not run on hardware
//
#include <hip/hip_runtime.h>

// ---------------------------------------------------------------------------
// Problem constants (reference: embeddings (8192, 512) fp32, THRESH=0.5)
// ---------------------------------------------------------------------------
#define N0 8192
#define DD 512
#define THRESH 0.5f
#define EPS 1e-8f
#define ROUNDS 14

// ctrl slots
#define C_N      0
#define C_ACTIVE 1
#define C_MERGE  2
#define C_NNEW   3

typedef __attribute__((ext_vector_type(16))) __bf16 v16bf;
typedef __attribute__((ext_vector_type(8)))  float  v8f;

union FragBF { v16bf v; unsigned int u[8]; };
union FragF  { v8f   v; float        f[8]; };

__device__ __forceinline__ unsigned short f2bf_rne(float f) {
    unsigned int u = __float_as_uint(f);
    unsigned int r = u + 0x7FFFu + ((u >> 16) & 1u);
    return (unsigned short)(r >> 16);
}

// ---------------------------------------------------------------------------
// init: copy input embeddings into working buffer, set control words
// ---------------------------------------------------------------------------
__global__ void mm_init_kernel(const float* __restrict__ in, float* __restrict__ embA,
                               int* __restrict__ ctrl) {
    long long idx = (long long)blockIdx.x * blockDim.x + threadIdx.x;
    if (idx < (long long)N0 * DD) embA[idx] = in[idx];
    if (blockIdx.x == 0 && threadIdx.x == 0) {
        ctrl[C_N] = N0; ctrl[C_ACTIVE] = 1; ctrl[C_MERGE] = 0; ctrl[C_NNEW] = N0;
    }
}

// ---------------------------------------------------------------------------
// prep: per-row L2 norm (clamped to EPS) + fp32 -> bf16 conversion
// ---------------------------------------------------------------------------
__global__ void mm_prep_kernel(const float* __restrict__ embA,
                               unsigned short* __restrict__ ebf,
                               float* __restrict__ norms,
                               const int* __restrict__ ctrl) {
    if (ctrl[C_ACTIVE] == 0) return;
    int n = ctrl[C_N];
    int row = blockIdx.x;
    if (row >= n) return;
    __shared__ float red[128];
    int t = threadIdx.x;
    const float* src = embA + (long long)row * DD;
    unsigned short* dst = ebf + (long long)row * DD;
    float s = 0.f;
    for (int e = t; e < DD; e += 128) {
        float v = src[e];
        s += v * v;
        dst[e] = f2bf_rne(v);
    }
    red[t] = s;
    __syncthreads();
    for (int w = 64; w > 0; w >>= 1) {
        if (t < w) red[t] += red[t + w];
        __syncthreads();
    }
    if (t == 0) norms[row] = fmaxf(sqrtf(red[0]), EPS);
}

// ---------------------------------------------------------------------------
// GEMM: sim = (E . E^T) / (norm_i * norm_j), bf16 WMMA, fp32 accumulate.
// Block = 128 threads = 4 waves; 64x64 macro-tile per block; each wave owns
// a 32x32 register tile = 2x2 WMMA tiles (4 accumulators). Per k-step of 32
// a wave loads 2 A-fragments + 2 B-fragments and issues 4 WMMAs (2x reuse).
// ---------------------------------------------------------------------------
__global__ __launch_bounds__(128)
void mm_gemm_kernel(const unsigned short* __restrict__ ebf,
                    const float* __restrict__ norms,
                    float* __restrict__ simF,
                    const int* __restrict__ ctrl) {
    if (ctrl[C_ACTIVE] == 0) return;
    int n = ctrl[C_N];
    int j0 = blockIdx.x * 64;
    int i0 = blockIdx.y * 64;
    if (i0 >= n || j0 >= n) return;
    if (j0 + 63 <= i0) return;               // only j > i is ever consumed

    __shared__ unsigned short As[64][40];    // 80B row pitch: 16B-aligned rows
    __shared__ unsigned short Bs[64][40];

    int t    = threadIdx.x;
    int lane = t & 31;
    int wave = t >> 5;                       // 0..3
    int wi = wave >> 1, wj = wave & 1;       // 2x2 quadrant of the 64x64 tile
    int half = lane >> 4;                    // lane group 0..1
    int m    = lane & 15;

    FragF acc00, acc01, acc10, acc11;
    acc00.v = (v8f){0.f,0.f,0.f,0.f,0.f,0.f,0.f,0.f};
    acc01.v = acc00.v; acc10.v = acc00.v; acc11.v = acc00.v;

    for (int kk = 0; kk < DD; kk += 32) {
        // stage A (rows i0..i0+63) and B (rows j0..j0+63), k-slice [kk, kk+32)
        // 64 rows x 32 ushorts = 256 uint4 per matrix; 2 per thread.
#pragma unroll
        for (int q = t; q < 256; q += 128) {
            int row = q >> 2, c4 = q & 3;
            *(uint4*)&As[row][c4 * 8] =
                *(const uint4*)(ebf + (long long)(i0 + row) * DD + kk + c4 * 8);
            *(uint4*)&Bs[row][c4 * 8] =
                *(const uint4*)(ebf + (long long)(j0 + row) * DD + kk + c4 * 8);
        }
        if (kk + 32 < DD) {                  // pull next k-chunk toward L2/L0
            __builtin_prefetch(ebf + (long long)(i0 + (t >> 1)) * DD + kk + 32, 0, 0);
            __builtin_prefetch(ebf + (long long)(j0 + (t >> 1)) * DD + kk + 32, 0, 0);
        }
        __syncthreads();

        // Build fragments per ISA 7.12.2 (16-bit A 16x32 / B 32x16 layouts)
        FragBF fa0, fa1, fb0, fb1;
        int ra = 32 * wi, rb = 32 * wj;
#pragma unroll
        for (int v = 0; v < 8; ++v) {
            int ka = 2 * v + ((v >= 4) ? 8 : 0) + (half ? 8 : 0);  // A k-pair
            fa0.u[v] = *(const unsigned int*)&As[ra + m][ka];
            fa1.u[v] = *(const unsigned int*)&As[ra + 16 + m][ka];
            int kb = 2 * v + ((v >= 4) ? 8 : 0) + (half ? 8 : 0);  // B k-pair
            fb0.u[v] = *(const unsigned int*)&Bs[rb + m][kb];
            fb1.u[v] = *(const unsigned int*)&Bs[rb + 16 + m][kb];
        }
        acc00.v = __builtin_amdgcn_wmma_f32_16x16x32_bf16(
            false, fa0.v, false, fb0.v, (short)0, acc00.v, false, false);
        acc01.v = __builtin_amdgcn_wmma_f32_16x16x32_bf16(
            false, fa0.v, false, fb1.v, (short)0, acc01.v, false, false);
        acc10.v = __builtin_amdgcn_wmma_f32_16x16x32_bf16(
            false, fa1.v, false, fb0.v, (short)0, acc10.v, false, false);
        acc11.v = __builtin_amdgcn_wmma_f32_16x16x32_bf16(
            false, fa1.v, false, fb1.v, (short)0, acc11.v, false, false);
        __syncthreads();
    }

    // store normalized similarity (C/D layout: VGPR p -> M = p + 8*half, N = m)
    FragF* accs[4] = { &acc00, &acc01, &acc10, &acc11 };
#pragma unroll
    for (int ti = 0; ti < 2; ++ti) {
#pragma unroll
        for (int tj = 0; tj < 2; ++tj) {
            int j = j0 + 32 * wj + 16 * tj + m;
            if (j >= n) continue;
            float nj = norms[j];
            FragF* a = accs[ti * 2 + tj];
#pragma unroll
            for (int p = 0; p < 8; ++p) {
                int i = i0 + 32 * wi + 16 * ti + p + (half ? 8 : 0);
                if (i < n) {
                    float ni = norms[i];
                    simF[(size_t)i * N0 + j] = a->f[p] / (ni * nj);
                }
            }
        }
    }
}

// ---------------------------------------------------------------------------
// scan: faithful sequential greedy matching + survivor compaction
// ---------------------------------------------------------------------------
__global__ __launch_bounds__(1024)
void mm_scan_kernel(const float* __restrict__ simF,
                    int* __restrict__ partners, int* __restrict__ consumed,
                    int* __restrict__ survivors, int* __restrict__ ctrl) {
    if (ctrl[C_ACTIVE] == 0) return;
    int n = ctrl[C_N];
    __shared__ char  merged[N0];
    __shared__ float sVal[1024];
    __shared__ int   sIdx[1024];
    int t = threadIdx.x;
    for (int i = t; i < N0; i += 1024) merged[i] = 0;
    __syncthreads();

    for (int i = 0; i < n; ++i) {
        const float* row = simF + (size_t)i * N0;
        float bv = -3.0e38f; int bi = -1;
        for (int j = i + 1 + t; j < n; j += 1024) {
            if (!merged[j]) {
                float v = row[j];
                if (v > bv) { bv = v; bi = j; }   // first-max within a thread
            }
        }
        sVal[t] = bv; sIdx[t] = bi;
        __syncthreads();
        for (int s = 512; s > 0; s >>= 1) {
            if (t < s) {
                float ov = sVal[t + s]; int oi = sIdx[t + s];
                if (ov > sVal[t] ||
                    (ov == sVal[t] && oi >= 0 && (sIdx[t] < 0 || oi < sIdx[t]))) {
                    sVal[t] = ov; sIdx[t] = oi;   // argmax ties -> lowest j
                }
            }
            __syncthreads();
        }
        if (t == 0) {
            int was = merged[i];
            int j = sIdx[0];
            int ok = (!was) && (j >= 0) && (sVal[0] >= THRESH);
            consumed[i] = was;
            partners[i] = ok ? j : -1;
            if (ok) { merged[i] = 1; merged[j] = 1; ctrl[C_MERGE] = 1; }
        }
        __syncthreads();
    }

    if (t == 0) {
        int cnt = 0;
        for (int i = 0; i < n; ++i)
            if (!consumed[i]) survivors[cnt++] = i;
        ctrl[C_NNEW] = cnt;
    }
}

// ---------------------------------------------------------------------------
// fuse + compact into the second buffer; then copy back
// ---------------------------------------------------------------------------
__global__ void mm_fuse_kernel(const float* __restrict__ cur, float* __restrict__ nxt,
                               const int* __restrict__ partners,
                               const int* __restrict__ survivors,
                               const int* __restrict__ ctrl) {
    if (ctrl[C_ACTIVE] == 0 || ctrl[C_MERGE] == 0) return;
    int s = blockIdx.x;
    if (s >= ctrl[C_NNEW]) return;
    int i = survivors[s];
    int p = partners[i];
    const float* a = cur + (long long)i * DD;
    const float* b = cur + (long long)(p >= 0 ? p : i) * DD;
    float* o = nxt + (long long)s * DD;
    for (int e = threadIdx.x; e < DD; e += blockDim.x) {
        float v = a[e];
        if (p >= 0) v = fminf(v + b[e], 1.0f);  // clamp(v1+v2, max=1.0)
        o[e] = v;
    }
}

__global__ void mm_copyback_kernel(const float* __restrict__ nxt, float* __restrict__ cur,
                                   const int* __restrict__ ctrl) {
    if (ctrl[C_ACTIVE] == 0 || ctrl[C_MERGE] == 0) return;
    int s = blockIdx.x;
    if (s >= ctrl[C_NNEW]) return;
    const float* a = nxt + (long long)s * DD;
    float* o = cur + (long long)s * DD;
    for (int e = threadIdx.x; e < DD; e += blockDim.x) o[e] = a[e];
}

__global__ void mm_epilogue_kernel(int* __restrict__ ctrl) {
    if (blockIdx.x == 0 && threadIdx.x == 0) {
        if (ctrl[C_ACTIVE]) {
            if (ctrl[C_MERGE]) {
                ctrl[C_N] = ctrl[C_NNEW];
                if (ctrl[C_N] <= 1) ctrl[C_ACTIVE] = 0;
            } else {
                ctrl[C_ACTIVE] = 0;            // no merge occurred -> break
            }
            ctrl[C_MERGE] = 0;
        }
    }
}

__global__ void mm_output_kernel(const float* __restrict__ embA, float* __restrict__ out,
                                 int out_elems) {
    long long idx = (long long)blockIdx.x * blockDim.x + threadIdx.x;
    if (idx < out_elems) out[idx] = embA[idx];
}

// ---------------------------------------------------------------------------
// launch
// ---------------------------------------------------------------------------
extern "C" void kernel_launch(void* const* d_in, const int* in_sizes, int n_in,
                              void* d_out, int out_size, void* d_ws, size_t ws_size,
                              hipStream_t stream) {
    (void)in_sizes; (void)n_in; (void)ws_size;
    const float* d_emb = (const float*)d_in[0];
    float* out = (float*)d_out;

    // workspace carve (all offsets 256B aligned)
    char* base = (char*)d_ws;
    size_t off = 0;
    float* simF = (float*)(base + off);                 off += (size_t)N0 * N0 * 4;  // 256 MB
    unsigned short* ebf = (unsigned short*)(base + off); off += (size_t)N0 * DD * 2; // 8 MB
    float* norms = (float*)(base + off);                off += (size_t)N0 * 4;
    int* partners = (int*)(base + off);                 off += (size_t)N0 * 4;
    int* consumed = (int*)(base + off);                 off += (size_t)N0 * 4;
    int* survivors = (int*)(base + off);                off += (size_t)N0 * 4;
    float* embA = (float*)(base + off);                 off += (size_t)N0 * DD * 4;  // 16 MB
    float* embB = (float*)(base + off);                 off += (size_t)N0 * DD * 4;  // 16 MB
    int* ctrl = (int*)(base + off);                     off += 256;

    {
        long long tot = (long long)N0 * DD;
        int blocks = (int)((tot + 255) / 256);
        mm_init_kernel<<<blocks, 256, 0, stream>>>(d_emb, embA, ctrl);
    }

    dim3 ggrid(N0 / 64, N0 / 64, 1);
    for (int r = 0; r < ROUNDS; ++r) {
        mm_prep_kernel<<<N0, 128, 0, stream>>>(embA, ebf, norms, ctrl);
        mm_gemm_kernel<<<ggrid, 128, 0, stream>>>(ebf, norms, simF, ctrl);
        mm_scan_kernel<<<1, 1024, 0, stream>>>(simF, partners, consumed, survivors, ctrl);
        mm_fuse_kernel<<<N0, 128, 0, stream>>>(embA, embB, partners, survivors, ctrl);
        mm_copyback_kernel<<<N0, 128, 0, stream>>>(embB, embA, ctrl);
        mm_epilogue_kernel<<<1, 1, 0, stream>>>(ctrl);
    }

    {
        int blocks = (out_size + 255) / 256;
        if (blocks < 1) blocks = 1;
        mm_output_kernel<<<blocks, 256, 0, stream>>>(embA, out, out_size);
    }
}
